// Model_82583631168036
// MI455X (gfx1250) — matrix operand order repaired and checked by reading, not compiled, not run on hardware
//
#include <hip/hip_runtime.h>
#include <math.h>

// ---------------------------------------------------------------------------
// Types for CDNA5 WMMA (wave32): v_wmma_f32_16x16x32_bf16
// ---------------------------------------------------------------------------
typedef __attribute__((ext_vector_type(16))) __bf16 bhalf16;
typedef __attribute__((ext_vector_type(2)))  __bf16 bhalf2;
typedef __attribute__((ext_vector_type(8)))  float  float8;

__device__ __forceinline__ float wred(float v) {
#pragma unroll
  for (int m = 16; m > 0; m >>= 1) v += __shfl_xor(v, m, 32);
  return v;
}
__device__ __forceinline__ float sigm_f(float x) { return 1.0f / (1.0f + __expf(-x)); }
__device__ __forceinline__ float gelu_f(float x) {
  float x3 = x * x * x;
  return 0.5f * x * (1.0f + tanhf(0.7978845608028654f * (x + 0.044715f * x3)));
}

// ---------------------------------------------------------------------------
// Software-pipelined bf16 WMMA GEMM, fused bias + activation.
//   C[m,n] = act( sum_k A[m,k]*B[k,n] + bias[n] )
// 256 threads = 8 wave32s, K-step 32. LDS tiles stored pre-swizzled in the ISA
// fragment layout; every per-thread LDS slot is loop-invariant. Next K-tile is
// prefetched into registers before the WMMA burst (VMEM latency overlaps the
// matrix pipe).
//
// Bounds strategy (no EXEC-mask divergence):
//  * M/N overhang: row/col indices clamped ONCE outside the K-loop; loads are
//    unconditional (duplicate data), results discarded in the epilogue.
//  * K overhang: only the single tail tile needs zeroing, and only on B
//    (A_garbage * 0 == 0). All full tiles: zero checks, zero selects.
// GUARD=false additionally assumes tile-exact M/N, K%32==0, csA==csB==1
// (float2 loads along K for A / N for B).
// ACT: 0=none 1=gelu(tanh) 2=sigmoid
// ---------------------------------------------------------------------------
template <int BM, int BN, int ACT, bool GUARD>
__launch_bounds__(256)
__global__ void gemm_wmma_kernel(const float* __restrict__ A,
                                 const float* __restrict__ B,
                                 const float* __restrict__ bias,
                                 float* __restrict__ C,
                                 int M, int N, int K,
                                 long long rsA, long long csA,
                                 long long rsB, long long csB,
                                 long long rsC,
                                 int inner,
                                 long long oA, long long iA,
                                 long long oB, long long iB,
                                 long long oC, long long iC) {
  constexpr int SM    = BM / 16;
  constexpr int SN    = BN / 16;
  constexpr int PER   = (SM * SN) / 8;  // subtiles per wave
  constexpr int AITER = BM / 16;        // A staging passes (k-pair per thread)
  constexpr int NPAIR = BN / 2;
  constexpr int KS    = 256 / NPAIR;    // B k-rows per staging pass
  constexpr int BITER = 32 / KS;

  __shared__ __align__(32) __bf16 lA[SM][32][16];
  __shared__ __align__(32) __bf16 lB[SN][32][16];

  const int z = blockIdx.z;
  const long long offA = (long long)(z / inner) * oA + (long long)(z % inner) * iA;
  const long long offB = (long long)(z / inner) * oB + (long long)(z % inner) * iB;
  const long long offC = (long long)(z / inner) * oC + (long long)(z % inner) * iC;

  const int tid  = threadIdx.x;
  const int lane = tid & 31;
  const int wave = tid >> 5;
  const int m0   = blockIdx.x * BM;
  const int n0   = blockIdx.y * BN;

  // thread staging coordinates
  const int akp = tid & 15, amb = tid >> 4;      // A: k = 2*akp, m = amb+16*i
  const int bnp = tid % NPAIR, bkb = tid / NPAIR;// B: n = 2*bnp, k = bkb+KS*i
  const int ak  = 2 * akp;

  // A row offsets, clamped once (k-loop invariant)
  long long aRow[AITER];
#pragma unroll
  for (int i = 0; i < AITER; ++i) {
    int gm = m0 + amb + 16 * i;
    if (GUARD) gm = gm < M ? gm : M - 1;
    aRow[i] = offA + (long long)gm * rsA;
  }
  // B column offsets (pair), clamped once
  int gn0 = n0 + 2 * bnp, gn1 = gn0 + 1;
  if (GUARD) { gn0 = gn0 < N ? gn0 : N - 1; gn1 = gn1 < N ? gn1 : N - 1; }
  const long long bCol0 = offB + (long long)gn0 * csB;
  const long long bCol1 = offB + (long long)gn1 * csB;

  // loop-invariant LDS slots
  const int ahalf = (ak >> 3) & 1;
  const int ae    = (ak & 7) | ((ak >> 4) << 3);
  const int alane = amb | (ahalf << 4);
  const int bsn   = (2 * bnp) >> 4;
  const int bnlo  = (2 * bnp) & 15;

  const float8 fzero = {0.f, 0.f, 0.f, 0.f, 0.f, 0.f, 0.f, 0.f};
  float8 acc[PER];
#pragma unroll
  for (int i = 0; i < PER; ++i) acc[i] = fzero;

  float2 ra[AITER], rb[BITER];

  auto loadTile = [&](int k0, bool tail) {
    if (!GUARD) {
#pragma unroll
      for (int i = 0; i < AITER; ++i)
        ra[i] = *reinterpret_cast<const float2*>(A + aRow[i] + (k0 + ak));
#pragma unroll
      for (int i = 0; i < BITER; ++i)
        rb[i] = *reinterpret_cast<const float2*>(
            B + (long long)(k0 + bkb + KS * i) * rsB + bCol0);
    } else if (!tail) {  // full tile: fully unconditional strided loads
#pragma unroll
      for (int i = 0; i < AITER; ++i) {
        ra[i].x = A[aRow[i] + (long long)(k0 + ak) * csA];
        ra[i].y = A[aRow[i] + (long long)(k0 + ak + 1) * csA];
      }
#pragma unroll
      for (int i = 0; i < BITER; ++i) {
        long long r = (long long)(k0 + bkb + KS * i) * rsB;
        rb[i].x = B[r + bCol0];
        rb[i].y = B[r + bCol1];
      }
    } else {             // tail tile: clamp k; zero only B's OOB-k rows
#pragma unroll
      for (int i = 0; i < AITER; ++i) {
        int k1 = k0 + ak;     k1 = k1 < K ? k1 : K - 1;
        int k2 = k0 + ak + 1; k2 = k2 < K ? k2 : K - 1;
        ra[i].x = A[aRow[i] + (long long)k1 * csA];
        ra[i].y = A[aRow[i] + (long long)k2 * csA];
      }
#pragma unroll
      for (int i = 0; i < BITER; ++i) {
        int k = k0 + bkb + KS * i;
        int kc = k < K ? k : K - 1;
        long long r = (long long)kc * rsB;
        float x = B[r + bCol0], y = B[r + bCol1];
        rb[i].x = k < K ? x : 0.f;
        rb[i].y = k < K ? y : 0.f;
      }
    }
  };

  auto storeLDS = [&]() {
#pragma unroll
    for (int i = 0; i < AITER; ++i) {
      bhalf2 p; p.x = (__bf16)ra[i].x; p.y = (__bf16)ra[i].y;
      *reinterpret_cast<bhalf2*>(&lA[i][alane][ae]) = p;
    }
#pragma unroll
    for (int i = 0; i < BITER; ++i) {
      int kl = bkb + KS * i;
      int ll = bnlo | (((kl >> 3) & 1) << 4);
      lB[bsn][ll][(kl & 7) | ((kl >> 4) << 3)]     = (__bf16)rb[i].x;
      lB[bsn][ll + 1][(kl & 7) | ((kl >> 4) << 3)] = (__bf16)rb[i].y;
    }
  };

  auto compute = [&]() {
    bhalf16 af = *reinterpret_cast<const bhalf16*>(&lA[0][lane][0]);
    int cur = 0;
#pragma unroll
    for (int i = 0; i < PER; ++i) {
      int st = wave * PER + i;
      int sm = st / SN, sn = st % SN;
      if (sm != cur) { af = *reinterpret_cast<const bhalf16*>(&lA[sm][lane][0]); cur = sm; }
      bhalf16 bf = *reinterpret_cast<const bhalf16*>(&lB[sn][lane][0]);
      acc[i] = __builtin_amdgcn_wmma_f32_16x16x32_bf16(
          false, af, false, bf, (short)0, acc[i], false, false);
    }
  };

  loadTile(0, GUARD && (K < 32));
  for (int t = 0;;) {
    storeLDS();
    __syncthreads();
    ++t;
    const bool more = 32 * t < K;
    if (more) loadTile(32 * t, GUARD && (32 * t + 32 > K));
    compute();
    __syncthreads();
    if (!more) break;
  }

  // --- epilogue: C/D layout VGPR r -> rows {r, r+8}, col = lane&15 ---------
#pragma unroll
  for (int i = 0; i < PER; ++i) {
    int st = wave * PER + i;
    int sm = st / SN, sn = st % SN;
    int col = n0 + sn * 16 + (lane & 15);
    if (GUARD && col >= N) continue;
    float bv = bias ? bias[col] : 0.f;
#pragma unroll
    for (int r = 0; r < 8; ++r) {
      int row = m0 + sm * 16 + r + ((lane >> 4) << 3);
      if (GUARD && row >= M) continue;
      float v = acc[i][r] + bv;
      if (ACT == 1) v = gelu_f(v);
      else if (ACT == 2) v = sigm_f(v);
      C[offC + (long long)row * rsC + col] = v;
    }
  }
}

// ---------------------------------------------------------------------------
// Elementwise / reduction kernels (fp32, bandwidth-trivial)
// ---------------------------------------------------------------------------
__global__ void add_kernel(float* x, const float* y, int n) {
  int i = blockIdx.x * 256 + threadIdx.x;
  if (i < n) x[i] += y[i];
}

// Per-row LayerNorm over D=512, one wave per row (in place).
__global__ void layernorm_kernel(float* x, const float* g, const float* b, int rows) {
  int row = blockIdx.x * 8 + (threadIdx.x >> 5);
  int lane = threadIdx.x & 31;
  if (row >= rows) return;
  float* xr = x + (long long)row * 512;
  float v[16]; float s = 0.f;
#pragma unroll
  for (int i = 0; i < 16; ++i) { v[i] = xr[lane + 32 * i]; s += v[i]; }
  s = wred(s) * (1.f / 512.f);
  float var = 0.f;
#pragma unroll
  for (int i = 0; i < 16; ++i) { float d = v[i] - s; var += d * d; }
  var = wred(var) * (1.f / 512.f);
  float rs = rsqrtf(var + 1e-5f);
#pragma unroll
  for (int i = 0; i < 16; ++i) { int c = lane + 32 * i; xr[c] = (v[i] - s) * rs * g[c] + b[c]; }
}

// out[b,h,e] = sum_s x[b,s,h*64+e] * (scale ? scale[b,s,h] : 1)   (grid B*H, 64 thr)
__global__ void colsum_kernel(const float* x, const float* scale, float* out, int S) {
  int bh = blockIdx.x; int b = bh >> 3, h = bh & 7; int e = threadIdx.x;
  const float* base = x + (long long)b * S * 512 + h * 64 + e;
  const float* sc = scale ? scale + (long long)b * S * 8 + h : nullptr;
  float s = 0.f;
  for (int t = 0; t < S; ++t) s += base[(long long)t * 512] * (sc ? sc[t * 8] : 1.f);
  out[bh * 64 + e] = s;
}

// out[b,s,h] = f( dot64(X[b,s,h,:], Y[b,h,:]) + 1e-6 )   grid=B*S, 256 thr (wave=h)
// MODE 0: reciprocal, 1: raw, 2: clip[1e-6,1e6]
template <int MODE>
__global__ void dot_bh_kernel(const float* X, const float* Y, float* out, int S) {
  int bs = blockIdx.x; int b = bs / S;
  int h = threadIdx.x >> 5, lane = threadIdx.x & 31;
  const float* xr = X + (long long)bs * 512 + h * 64;
  const float* yr = Y + ((long long)b * 8 + h) * 64;
  float a = xr[lane] * yr[lane] + xr[lane + 32] * yr[lane + 32];
  a = wred(a);
  if (lane == 0) {
    float d = a + 1e-6f;
    if (MODE == 0) d = 1.f / d;
    if (MODE == 2) d = fminf(fmaxf(d, 1e-6f), 1e6f);
    out[bs * 8 + h] = d;
  }
}

// out[b,h] = sum_s x[b,s,h]   (grid B*H, 32 thr)
__global__ void sums_kernel(const float* x, float* out, int S) {
  int bh = blockIdx.x; int b = bh >> 3, h = bh & 7; int lane = threadIdx.x;
  float s = 0.f;
  for (int t = lane; t < S; t += 32) s += x[((long long)b * S + t) * 8 + h];
  s = wred(s);
  if (lane == 0) out[bh] = s;
}

// k *= source_comp = ccs * S / css_sum   (softmax(log c) * S == c/sum(c) * S)
__global__ void scale_src_kernel(float* k, const float* ccs, const float* css_sum,
                                 int S, int total) {
  int idx = blockIdx.x * 256 + threadIdx.x;
  if (idx >= total) return;
  int d = idx & 511; int bs = idx >> 9; int h = d >> 6; int b = bs / S;
  k[idx] *= ccs[bs * 8 + h] * (float)S / css_sum[b * 8 + h];
}

// attn *= sink_in * sigmoid(cs * S / cs_sum)
__global__ void attn_scale_kernel(float* a, const float* sink_in, const float* cs,
                                  const float* cs_sum, int S, int total) {
  int idx = blockIdx.x * 256 + threadIdx.x;
  if (idx >= total) return;
  int d = idx & 511; int bs = idx >> 9; int h = d >> 6; int b = bs / S;
  float alloc = sigm_f(cs[bs * 8 + h] * (float)S / cs_sum[b * 8 + h]);
  a[idx] *= sink_in[bs * 8 + h] * alloc;
}

// e rows b*2:   += pe[0] (sin(0)=0 even dims, cos(0)=1 odd dims)
// e rows b*2+1: = glb broadcast
__global__ void build_e_kernel(float* e, const float* glb) {
  int idx = blockIdx.x * 256 + threadIdx.x;
  if (idx >= 64 * 512) return;
  int b = idx >> 9, d = idx & 511;
  e[((long long)b * 2) * 512 + d] += (d & 1) ? 1.0f : 0.0f;
  e[((long long)b * 2 + 1) * 512 + d] = glb[d];
}

// endo (B,384) zero-padded: endo[b,t] = x_enc[b,t,0] for t<365, else 0
__global__ void endo_pad_kernel(const float* x_enc, float* endo) {
  int idx = blockIdx.x * 256 + threadIdx.x;
  if (idx >= 64 * 384) return;
  int t = idx % 384, b = idx / 384;
  endo[idx] = (t < 365) ? x_enc[(long long)b * 365 * 39 + (long long)t * 39] : 0.f;
}

// Forward/backward zero-fill over channels 21..38 of x_enc -> xi (B,365,18)
__global__ void fill_kernel(const float* x_enc, float* xi) {
  int id = blockIdx.x * 64 + threadIdx.x;
  if (id >= 64 * 18) return;
  int b = id / 18, n = id % 18;
  const float* x = x_enc + (long long)b * 365 * 39 + (21 + n);
  float* o = xi + ((long long)b * 365) * 18 + n;
  float carry = x[0];
  for (int t = 0; t < 365; ++t) {
    float xv = x[(long long)t * 39];
    float v = (t == 0) ? xv : ((xv == 0.f) ? carry : xv);
    carry = v;
    o[(long long)t * 18] = v;
  }
  carry = x[364LL * 39];
  for (int t = 364; t >= 0; --t) {
    float xv = x[(long long)t * 39];
    float v = (t == 364) ? xv : ((xv == 0.f) ? carry : xv);
    carry = v;
    o[(long long)t * 18] = 0.5f * (o[(long long)t * 18] + v);
  }
}

// rbf = mean_j exp(-10 (xi - c_j)^2), c_j = linspace(-1,1,50)
__global__ void rbf_kernel(const float* xi, float* rbf, int total) {
  int idx = blockIdx.x * 256 + threadIdx.x;
  if (idx >= total) return;
  float v = xi[idx], s = 0.f;
#pragma unroll
  for (int j = 0; j < 50; ++j) {
    float c = -1.f + (2.f / 49.f) * j;
    float d = v - c;
    s += __expf(-10.f * d * d);
  }
  rbf[idx] = s * (1.f / 50.f);
}

// wavelet features collapse to endpoint reads:
// d1 = (x[364]-x[0])/364 ; d2 = (x[363]+x[364]-x[0]-x[1])/363
__global__ void wavelet_kernel(const float* x_enc, const float* rbf, float* wcomb) {
  int id = blockIdx.x * 64 + threadIdx.x;
  if (id >= 64 * 30) return;
  int b = id / 30, c = id % 30;
  float x0, x1, x363, x364;
  if (c < 12) {  // weather = x_enc channel 2+c
    const float* x = x_enc + (long long)b * 365 * 39 + (2 + c);
    x0 = x[0]; x1 = x[39]; x363 = x[363LL * 39]; x364 = x[364LL * 39];
  } else {       // modis rbf channel c-12
    const float* x = rbf + (long long)b * 365 * 18 + (c - 12);
    x0 = x[0]; x1 = x[18]; x363 = x[363LL * 18]; x364 = x[364LL * 18];
  }
  wcomb[b * 60 + c]      = (x364 - x0) * (1.f / 364.f);
  wcomb[b * 60 + 30 + c] = (x363 + x364 - x0 - x1) * (1.f / 363.f);
}

// inv (B,105,384) zero-padded in t: rows = [exo2(38ch)^T, w_exp(60ch), x_mark(7ch)]
__global__ void inv_kernel(const float* x_enc, const float* rbf, const float* wcomb,
                           const float* xmark, float* inv, int total) {
  int idx = blockIdx.x * 256 + threadIdx.x;
  if (idx >= total) return;
  int t = idx % 384; int c = (idx / 384) % 105; int b = idx / (384 * 105);
  float v = 0.f;
  if (t < 365) {
    if (c < 20)      v = x_enc[((long long)b * 365 + t) * 39 + 1 + c];
    else if (c < 38) v = rbf[((long long)b * 365 + t) * 18 + (c - 20)];
    else if (c < 98) v = wcomb[b * 60 + (c - 38)];
    else             v = xmark[((long long)b * 365 + t) * 7 + (c - 98)];
  }
  inv[idx] = v;
}

// enc (B,106,512): token0 = endo_embed (e row b*2), tokens 1..105 = tmp
__global__ void build_enc_kernel(const float* efin, const float* tmp, float* enc) {
  int idx = blockIdx.x * 256 + threadIdx.x;
  if (idx >= 64 * 106 * 512) return;
  int d = idx & 511; int tok = (idx >> 9) % 106; int b = idx / (106 * 512);
  enc[idx] = (tok == 0) ? efin[((long long)b * 2) * 512 + d]
                        : tmp[((long long)b * 105 + (tok - 1)) * 512 + d];
}

// out[b,p,n] = dot(enc[b,n,:], proj_w[:,p]) + proj_b[p], tokens n<39, transposed
__global__ void proj_out_kernel(const float* enc, const float* w, const float* bias,
                                float* out) {
  int id = blockIdx.x * 8 + (threadIdx.x >> 5);
  int lane = threadIdx.x & 31;
  if (id >= 64 * 7 * 39) return;
  int n = id % 39; int p = (id / 39) % 7; int b = id / (39 * 7);
  const float* xr = enc + ((long long)b * 106 + n) * 512;
  float s = 0.f;
#pragma unroll
  for (int i = 0; i < 16; ++i) { int c = lane + 32 * i; s += xr[c] * w[c * 7 + p]; }
  s = wred(s);
  if (lane == 0) out[(long long)b * 7 * 39 + p * 39 + n] = s + bias[p];
}

// ---------------------------------------------------------------------------
// Host-side orchestration
// ---------------------------------------------------------------------------
static void launch_gemm(hipStream_t s, const float* A, const float* B, const float* bias,
                        float* C, int M, int N, int K,
                        long long rsA, long long csA, long long rsB, long long csB,
                        long long rsC, int act, int batch, int inner,
                        long long oA, long long iA, long long oB, long long iB,
                        long long oC, long long iC, int tile) {
  dim3 blk(256);
  if (tile == 64) {
    dim3 g((M + 63) / 64, (N + 63) / 64, batch);
    const bool fast = (csA == 1) && (csB == 1) && (K % 32 == 0) &&
                      (M % 64 == 0) && (N % 64 == 0);
    if (fast)
      gemm_wmma_kernel<64, 64, 0, false><<<g, blk, 0, s>>>(
          A, B, bias, C, M, N, K, rsA, csA, rsB, csB, rsC, inner, oA, iA, oB, iB, oC, iC);
    else
      gemm_wmma_kernel<64, 64, 0, true><<<g, blk, 0, s>>>(
          A, B, bias, C, M, N, K, rsA, csA, rsB, csB, rsC, inner, oA, iA, oB, iB, oC, iC);
    return;
  }
  dim3 g((M + 127) / 128, (N + 127) / 128, batch);
  const bool fast = (csA == 1) && (csB == 1) && (K % 32 == 0) &&
                    (M % 128 == 0) && (N % 128 == 0);
  if (fast) {
    if (act == 0)
      gemm_wmma_kernel<128, 128, 0, false><<<g, blk, 0, s>>>(
          A, B, bias, C, M, N, K, rsA, csA, rsB, csB, rsC, inner, oA, iA, oB, iB, oC, iC);
    else if (act == 1)
      gemm_wmma_kernel<128, 128, 1, false><<<g, blk, 0, s>>>(
          A, B, bias, C, M, N, K, rsA, csA, rsB, csB, rsC, inner, oA, iA, oB, iB, oC, iC);
    else
      gemm_wmma_kernel<128, 128, 2, false><<<g, blk, 0, s>>>(
          A, B, bias, C, M, N, K, rsA, csA, rsB, csB, rsC, inner, oA, iA, oB, iB, oC, iC);
  } else {
    if (act == 0)
      gemm_wmma_kernel<128, 128, 0, true><<<g, blk, 0, s>>>(
          A, B, bias, C, M, N, K, rsA, csA, rsB, csB, rsC, inner, oA, iA, oB, iB, oC, iC);
    else if (act == 1)
      gemm_wmma_kernel<128, 128, 1, true><<<g, blk, 0, s>>>(
          A, B, bias, C, M, N, K, rsA, csA, rsB, csB, rsC, inner, oA, iA, oB, iB, oC, iC);
    else
      gemm_wmma_kernel<128, 128, 2, true><<<g, blk, 0, s>>>(
          A, B, bias, C, M, N, K, rsA, csA, rsB, csB, rsC, inner, oA, iA, oB, iB, oC, iC);
  }
}

struct LayerP {
  const float *b1, *b2, *bk, *bo, *bq, *bv;
  const float *ln1b, *ln1g, *ln2b, *ln2g;
  const float *w1, *w2, *wk, *wo, *wq, *wv;
};
static LayerP load_layer(void* const* d_in, int base) {
  LayerP p;
  p.b1 = (const float*)d_in[base + 0];  p.b2 = (const float*)d_in[base + 1];
  p.bk = (const float*)d_in[base + 2];  p.bo = (const float*)d_in[base + 3];
  p.bq = (const float*)d_in[base + 4];  p.bv = (const float*)d_in[base + 5];
  p.ln1b = (const float*)d_in[base + 6]; p.ln1g = (const float*)d_in[base + 7];
  p.ln2b = (const float*)d_in[base + 8]; p.ln2g = (const float*)d_in[base + 9];
  p.w1 = (const float*)d_in[base + 10]; p.w2 = (const float*)d_in[base + 11];
  p.wk = (const float*)d_in[base + 12]; p.wo = (const float*)d_in[base + 13];
  p.wq = (const float*)d_in[base + 14]; p.wv = (const float*)d_in[base + 15];
  return p;
}

struct AttnWS {
  float *q, *k, *v, *attn, *t1, *y, *kv;
  float *sink_in, *source_out, *cs, *ccs;
  float *ksum, *qsum, *kso, *qsi, *cs_sum, *css_sum;
};

static void run_encoder_layer(hipStream_t s, float* x, int Bb, int S,
                              const LayerP& P, const AttnWS& W) {
  const int M = Bb * S;
  const int EW = (M * 512 + 255) / 256;
  // Q/K/V projections (sigmoid fused for q,k — flow attention applies it)
  launch_gemm(s, x, P.wq, P.bq, W.q, M, 512, 512, 512, 1, 512, 1, 512, 2, 1, 1, 0,0,0,0,0,0, 128);
  launch_gemm(s, x, P.wk, P.bk, W.k, M, 512, 512, 512, 1, 512, 1, 512, 2, 1, 1, 0,0,0,0,0,0, 128);
  launch_gemm(s, x, P.wv, P.bv, W.v, M, 512, 512, 512, 1, 512, 1, 512, 0, 1, 1, 0,0,0,0,0,0, 128);
  // flow attention bookkeeping
  colsum_kernel<<<Bb * 8, 64, 0, s>>>(W.k, nullptr, W.ksum, S);
  colsum_kernel<<<Bb * 8, 64, 0, s>>>(W.q, nullptr, W.qsum, S);
  dot_bh_kernel<0><<<M, 256, 0, s>>>(W.q, W.ksum, W.sink_in, S);
  dot_bh_kernel<0><<<M, 256, 0, s>>>(W.k, W.qsum, W.source_out, S);
  colsum_kernel<<<Bb * 8, 64, 0, s>>>(W.k, W.source_out, W.kso, S);
  colsum_kernel<<<Bb * 8, 64, 0, s>>>(W.q, W.sink_in, W.qsi, S);
  dot_bh_kernel<1><<<M, 256, 0, s>>>(W.q, W.kso, W.cs, S);
  dot_bh_kernel<2><<<M, 256, 0, s>>>(W.k, W.qsi, W.ccs, S);
  sums_kernel<<<Bb * 8, 32, 0, s>>>(W.cs, W.cs_sum, S);
  sums_kernel<<<Bb * 8, 32, 0, s>>>(W.ccs, W.css_sum, S);
  scale_src_kernel<<<EW, 256, 0, s>>>(W.k, W.ccs, W.css_sum, S, M * 512);
  // kv[b,h] = k_scaled^T (64xS) @ v (Sx64), batched WMMA over B*H
  launch_gemm(s, W.k, W.v, nullptr, W.kv, 64, 64, S,
              1, 512, 512, 1, 64, 0, Bb * 8, 8,
              (long long)S * 512, 64, (long long)S * 512, 64,
              8LL * 4096, 4096, 64);
  // attn[b,:,h] = q (Sx64) @ kv (64x64), batched WMMA
  launch_gemm(s, W.q, W.kv, nullptr, W.attn, S, 64, 64,
              512, 1, 64, 1, 512, 0, Bb * 8, 8,
              (long long)S * 512, 64, 8LL * 4096, 4096,
              (long long)S * 512, 64, 64);
  attn_scale_kernel<<<EW, 256, 0, s>>>(W.attn, W.sink_in, W.cs, W.cs_sum, S, M * 512);
  // output projection + residual + LN1
  launch_gemm(s, W.attn, P.wo, P.bo, W.t1, M, 512, 512, 512, 1, 512, 1, 512, 0, 1, 1, 0,0,0,0,0,0, 128);
  add_kernel<<<EW, 256, 0, s>>>(x, W.t1, M * 512);
  layernorm_kernel<<<(M + 7) / 8, 256, 0, s>>>(x, P.ln1g, P.ln1b, M);
  // FFN (GELU fused) + residual + LN2
  launch_gemm(s, x, P.w1, P.b1, W.y, M, 2048, 512, 512, 1, 2048, 1, 2048, 1, 1, 1, 0,0,0,0,0,0, 128);
  launch_gemm(s, W.y, P.w2, P.b2, W.t1, M, 512, 2048, 2048, 1, 512, 1, 512, 0, 1, 1, 0,0,0,0,0,0, 128);
  add_kernel<<<EW, 256, 0, s>>>(x, W.t1, M * 512);
  layernorm_kernel<<<(M + 7) / 8, 256, 0, s>>>(x, P.ln2g, P.ln2b, M);
}

static inline float* bump(char*& p, size_t nfloats) {
  float* r = (float*)p;
  p += ((nfloats * sizeof(float) + 255) / 256) * 256;
  return r;
}

extern "C" void kernel_launch(void* const* d_in, const int* in_sizes, int n_in,
                              void* d_out, int out_size, void* d_ws, size_t ws_size,
                              hipStream_t stream) {
  (void)in_sizes; (void)n_in; (void)out_size; (void)ws_size;
  const float* x_enc  = (const float*)d_in[0];
  const float* x_mark = (const float*)d_in[1];
  // params flattened alphabetically after the 4 data inputs:
  LayerP enL = load_layer(d_in, 4);
  const float* en_norm_b = (const float*)d_in[20];
  const float* en_norm_g = (const float*)d_in[21];
  const float* en_val_w  = (const float*)d_in[22];
  const float* enc_emb_b = (const float*)d_in[23];
  const float* enc_emb_w = (const float*)d_in[24];
  const float* glb       = (const float*)d_in[25];
  LayerP L0 = load_layer(d_in, 26);
  LayerP L1 = load_layer(d_in, 42);
  const float* norm_b = (const float*)d_in[58];
  const float* norm_g = (const float*)d_in[59];
  const float* proj_b = (const float*)d_in[60];
  const float* proj_w = (const float*)d_in[61];

  const int Mmax = 64 * 106;
  char* p = (char*)d_ws;
  AttnWS W;
  W.q    = bump(p, (size_t)Mmax * 512);
  W.k    = bump(p, (size_t)Mmax * 512);
  W.v    = bump(p, (size_t)Mmax * 512);
  W.attn = bump(p, (size_t)Mmax * 512);
  W.t1   = bump(p, (size_t)Mmax * 512);
  W.y    = bump(p, (size_t)Mmax * 2048);
  W.kv   = bump(p, 512 * 4096);
  W.sink_in    = bump(p, (size_t)Mmax * 8);
  W.source_out = bump(p, (size_t)Mmax * 8);
  W.cs         = bump(p, (size_t)Mmax * 8);
  W.ccs        = bump(p, (size_t)Mmax * 8);
  W.ksum   = bump(p, 512 * 64);
  W.qsum   = bump(p, 512 * 64);
  W.kso    = bump(p, 512 * 64);
  W.qsi    = bump(p, 512 * 64);
  W.cs_sum  = bump(p, 512);
  W.css_sum = bump(p, 512);
  float* e     = bump(p, 128 * 512);
  float* enc   = bump(p, (size_t)Mmax * 512);
  float* wcomb = bump(p, 64 * 60);
  float* endo  = bump(p, 64 * 384);
  // staging buffers alias large workspace regions not live at that point:
  float* xi     = W.q;     // (64,365,18) during feature stage
  float* rbfb   = W.k;     // (64,365,18)
  float* invb   = W.y;     // (64,105,384) padded
  float* tmpenc = W.attn;  // (64*105,512)

  // --- Stage A: endogenous encoder (B=64, S=2, D=512) ----------------------
  // endo materialized zero-padded to K=384 -> tile-exact fast GEMM (64x64)
  endo_pad_kernel<<<(64 * 384 + 255) / 256, 256, 0, stream>>>(x_enc, endo);
  launch_gemm(stream, endo, en_val_w, nullptr, e, 64, 512, 384,
              384, 1, 512, 1, 1024, 0, 1, 1, 0,0,0,0,0,0, 64);
  build_e_kernel<<<(64 * 512 + 255) / 256, 256, 0, stream>>>(e, glb);
  run_encoder_layer(stream, e, 64, 2, enL, W);
  layernorm_kernel<<<(128 + 7) / 8, 256, 0, stream>>>(e, en_norm_g, en_norm_b, 128);

  // --- Stage B: exogenous feature engineering ------------------------------
  fill_kernel<<<(64 * 18 + 63) / 64, 64, 0, stream>>>(x_enc, xi);
  rbf_kernel<<<(64 * 365 * 18 + 255) / 256, 256, 0, stream>>>(xi, rbfb, 64 * 365 * 18);
  wavelet_kernel<<<(64 * 30 + 63) / 64, 64, 0, stream>>>(x_enc, rbfb, wcomb);
  inv_kernel<<<(64 * 105 * 384 + 255) / 256, 256, 0, stream>>>(
      x_enc, rbfb, wcomb, x_mark, invb, 64 * 105 * 384);
  // M=6720=105*64, K=384, N=512 -> tile-exact fast GEMM (64x64)
  launch_gemm(stream, invb, enc_emb_w, enc_emb_b, tmpenc, 64 * 105, 512, 384,
              384, 1, 512, 1, 512, 0, 1, 1, 0,0,0,0,0,0, 64);
  build_enc_kernel<<<(64 * 106 * 512 + 255) / 256, 256, 0, stream>>>(e, tmpenc, enc);

  // --- Stage C: main encoder (B=64, S=106), 2 layers -----------------------
  run_encoder_layer(stream, enc, 64, 106, L0, W);
  run_encoder_layer(stream, enc, 64, 106, L1, W);
  layernorm_kernel<<<(64 * 106 + 7) / 8, 256, 0, stream>>>(enc, norm_g, norm_b, 64 * 106);

  // --- Projection + transpose + slice to (64,7,39) -------------------------
  proj_out_kernel<<<(64 * 7 * 39 + 7) / 8, 256, 0, stream>>>(
      enc, proj_w, proj_b, (float*)d_out);
}
